// UIC_29789893165643
// MI455X (gfx1250) — matrix operand order repaired and checked by reading, not compiled, not run on hardware
//
#include <hip/hip_runtime.h>
#include <hip/hip_bf16.h>
#include <math.h>

// ---------------------------------------------------------------------------
// CDNA5 / gfx1250: wave32, WMMA 16x16x32 bf16 (f32 accumulate)
// ---------------------------------------------------------------------------
typedef __attribute__((ext_vector_type(16))) __bf16 v16bf;
typedef __attribute__((ext_vector_type(8)))  float  v8f;

union BF16x16 { v16bf v; uint4 q[2]; };

// Problem constants (fixed by the reference)
constexpr int Bc = 16;     // batch
constexpr int Lc = 4096;   // sequence
constexpr int Dc = 512;    // input dim
constexpr int E1 = 256;    // enc1 dim
constexpr int E2 = 64;     // latent dim
constexpr int Hc = 511;    // (L-16)/8+1 patches
constexpr int NP = Bc * Hc;        // 8176 patch rows
constexpr int Mrows = Bc * Lc;     // 65536 token rows

// ---------------------------------------------------------------------------
// WMMA GEMM: C[M,N] = act(A[M,K] @ W[K,N] + bias), W pre-transposed (N,K) bf16.
// Each wave computes a 16x64 tile: ONE A fragment feeds FOUR WMMAs per K-step
// (4x A-operand reuse, 4 wmma per 6+ loads). 4 waves / block.
// A-frag (16x32 bf16): lane m=l&15, hi=l>>4; t<8 -> k=hi*8+t, t>=8 -> 16+hi*8+(t-8)
// B-frag (32x16 bf16): lane n=l&15, hi=l>>4; t<8 -> k=hi*8+t, t>=8 -> 16+hi*8+(t-8)
// C/D   (16x16 f32):   lane n=l&15; VGPR r -> m = r + 8*hi
// ---------------------------------------------------------------------------
template <typename AT, typename OT, int ACT>
__global__ __launch_bounds__(128) void gemm_wmma(
    const AT* __restrict__ A, const __bf16* __restrict__ Wt,
    const float* __restrict__ bias, OT* __restrict__ C,
    int M, int N, int K)
{
  const int lane = threadIdx.x & 31;
  const int wid  = threadIdx.x >> 5;
  const int colGroups = N >> 6;                  // 64-column groups
  const int tile = blockIdx.x * 4 + wid;
  if (tile >= (M >> 4) * colGroups) return;      // whole wave exits together
  const int tm = tile / colGroups;
  const int tc = tile - tm * colGroups;
  const int half = lane >> 4;
  const int l16  = lane & 15;
  const int m  = tm * 16 + l16;
  const int n0 = tc * 64 + l16;

  const AT*     __restrict__ arow = A + (size_t)m * K;
  const __bf16* __restrict__ bp0 = Wt + (size_t)(n0     ) * K;
  const __bf16* __restrict__ bp1 = Wt + (size_t)(n0 + 16) * K;
  const __bf16* __restrict__ bp2 = Wt + (size_t)(n0 + 32) * K;
  const __bf16* __restrict__ bp3 = Wt + (size_t)(n0 + 48) * K;

  v8f acc0, acc1, acc2, acc3;
#pragma unroll
  for (int r = 0; r < 8; ++r) { acc0[r] = 0.f; acc1[r] = 0.f; acc2[r] = 0.f; acc3[r] = 0.f; }

  for (int kk = 0; kk < K; kk += 32) {
    const int k0 = kk + half * 8;
    v16bf a;
    if constexpr (__is_same(AT, float)) {
      const float* p0 = arow + k0;
      const float* p1 = arow + k0 + 16;
#pragma unroll
      for (int t = 0; t < 8; ++t) {
        a[t]     = (__bf16)p0[t];
        a[t + 8] = (__bf16)p1[t];
      }
    } else {
      BF16x16 ta;
      ta.q[0] = *(const uint4*)(arow + k0);       // 8 bf16, 16B aligned
      ta.q[1] = *(const uint4*)(arow + k0 + 16);
      a = ta.v;
    }
    const int bo = kk + half * 8;                 // 2x 8 bf16 at bo and bo+16, 16B
    BF16x16 tb0, tb1, tb2, tb3;
    tb0.q[0] = *(const uint4*)(bp0 + bo); tb0.q[1] = *(const uint4*)(bp0 + bo + 16);
    tb1.q[0] = *(const uint4*)(bp1 + bo); tb1.q[1] = *(const uint4*)(bp1 + bo + 16);
    tb2.q[0] = *(const uint4*)(bp2 + bo); tb2.q[1] = *(const uint4*)(bp2 + bo + 16);
    tb3.q[0] = *(const uint4*)(bp3 + bo); tb3.q[1] = *(const uint4*)(bp3 + bo + 16);
    if (kk + 32 < K) __builtin_prefetch(arow + kk + 32, 0, 3);  // WGP-scope
    acc0 = __builtin_amdgcn_wmma_f32_16x16x32_bf16(false, a, false, tb0.v, (short)0, acc0, false, false);
    acc1 = __builtin_amdgcn_wmma_f32_16x16x32_bf16(false, a, false, tb1.v, (short)0, acc1, false, false);
    acc2 = __builtin_amdgcn_wmma_f32_16x16x32_bf16(false, a, false, tb2.v, (short)0, acc2, false, false);
    acc3 = __builtin_amdgcn_wmma_f32_16x16x32_bf16(false, a, false, tb3.v, (short)0, acc3, false, false);
  }

  float bv0 = 0.f, bv1 = 0.f, bv2 = 0.f, bv3 = 0.f;
  if (bias) { bv0 = bias[n0]; bv1 = bias[n0 + 16]; bv2 = bias[n0 + 32]; bv3 = bias[n0 + 48]; }
#pragma unroll
  for (int r = 0; r < 8; ++r) {
    const int mr = tm * 16 + r + half * 8;
    float x0 = acc0[r] + bv0, x1 = acc1[r] + bv1, x2 = acc2[r] + bv2, x3 = acc3[r] + bv3;
    if (ACT == 1) {
      x0 = x0 > 0.f ? x0 : 0.01f * x0;
      x1 = x1 > 0.f ? x1 : 0.01f * x1;
      x2 = x2 > 0.f ? x2 : 0.01f * x2;
      x3 = x3 > 0.f ? x3 : 0.01f * x3;
    }
    OT* crow = C + (size_t)mr * N + n0;
    if constexpr (__is_same(OT, __bf16)) {
      crow[0] = (__bf16)x0; crow[16] = (__bf16)x1; crow[32] = (__bf16)x2; crow[48] = (__bf16)x3;
    } else {
      crow[0] = x0; crow[16] = x1; crow[32] = x2; crow[48] = x3;
    }
  }
}

// Convert f32 weight (K,N) -> bf16 transposed (N,K)
__global__ void cvt_transpose(const float* __restrict__ W,
                              __bf16* __restrict__ Wt, int K, int N)
{
  int idx = blockIdx.x * blockDim.x + threadIdx.x;
  if (idx >= K * N) return;
  int k = idx / N, n = idx - k * N;
  Wt[(size_t)n * K + k] = (__bf16)W[idx];
}

// per_patch[b*Hc+i, d] = mean over 16 rows of Xe[b, i*8 + p, d]
__global__ void patch_mean(const float* __restrict__ Xe, float* __restrict__ P)
{
  int idx = blockIdx.x * blockDim.x + threadIdx.x;
  if (idx >= NP * E2) return;
  int d  = idx & 63;
  int bi = idx >> 6;
  int i  = bi % Hc;
  int b  = bi / Hc;
  const float* base = Xe + ((size_t)b * Lc + i * 8) * E2 + d;
  float s = 0.f;
#pragma unroll
  for (int p = 0; p < 16; ++p) s += base[(size_t)p * E2];
  P[idx] = s * (1.0f / 16.0f);
}

// in-place L2 row normalization with +eps
__global__ void rownorm(float* __restrict__ Q, int rows)
{
  int r = blockIdx.x * blockDim.x + threadIdx.x;
  if (r >= rows) return;
  float* p = Q + (size_t)r * E2;
  float s = 0.f;
  for (int d = 0; d < E2; ++d) s += p[d] * p[d];
  float inv = 1.0f / (sqrtf(s) + 1e-8f);
  for (int d = 0; d < E2; ++d) p[d] *= inv;
}

// One wave per patch: A = relu(tanh(qn_i kn_j - kn_i qn_j)), top-8 per row,
// renormalize over selected, Av_i = sum_j A~_ij v_j. Lane l owns j=l and j=l+32.
__global__ void attention_topk(const float* __restrict__ Qn,
                               const float* __restrict__ Kn,
                               const float* __restrict__ V,
                               float* __restrict__ Av, int rows)
{
  int wave = (blockIdx.x * blockDim.x + threadIdx.x) >> 5;
  int lane = threadIdx.x & 31;
  if (wave >= rows) return;
  const float* q = Qn + (size_t)wave * E2;
  const float* k = Kn + (size_t)wave * E2;
  const float* v = V  + (size_t)wave * E2;
  float q0 = q[lane], q1 = q[lane + 32];
  float k0 = k[lane], k1 = k[lane + 32];
  float v0 = v[lane], v1 = v[lane + 32];

  for (int i = 0; i < 64; ++i) {
    float qi = __shfl((i & 32) ? q1 : q0, i & 31, 32);
    float ki = __shfl((i & 32) ? k1 : k0, i & 31, 32);
    float a0 = tanhf(qi * k0 - ki * q0); a0 = a0 > 0.f ? a0 : 0.f;
    float a1 = tanhf(qi * k1 - ki * q1); a1 = a1 > 0.f ? a1 : 0.f;
    float sa = 0.f, sav = 0.f;
#pragma unroll
    for (int t = 0; t < 8; ++t) {
      float bvv; int bidx;
      if (a0 >= a1) { bvv = a0; bidx = lane; }            // lower index on tie
      else          { bvv = a1; bidx = lane + 32; }
#pragma unroll
      for (int off = 16; off > 0; off >>= 1) {
        float ov = __shfl_xor(bvv, off, 32);
        int   oi = __shfl_xor(bidx, off, 32);
        if (ov > bvv || (ov == bvv && oi < bidx)) { bvv = ov; bidx = oi; }
      }
      sa += bvv;
      float vv = __shfl((bidx & 32) ? v1 : v0, bidx & 31, 32);
      sav += bvv * vv;
      if (lane == (bidx & 31)) { if (bidx & 32) a1 = -1.f; else a0 = -1.f; }
    }
    if (lane == 0) Av[(size_t)wave * E2 + i] = sav / fmaxf(sa, 1e-8f);
  }
}

// out[b,l,d] = mean over patches i covering l of z[b,i,d]
__global__ void patch_scatter(const float* __restrict__ Z, float* __restrict__ O)
{
  int idx = blockIdx.x * blockDim.x + threadIdx.x;
  if (idx >= Mrows * E2) return;
  int d  = idx & 63;
  int bl = idx >> 6;
  int l  = bl & (Lc - 1);
  int b  = bl >> 12;                 // Lc = 4096 = 2^12
  int i1 = l >> 3; if (i1 > Hc - 1) i1 = Hc - 1;
  int t  = l - 15;
  int i0 = t <= 0 ? 0 : ((t + 7) >> 3);
  float s = 0.f; int cnt = 0;
  for (int i = i0; i <= i1; ++i) { s += Z[((size_t)b * Hc + i) * E2 + d]; ++cnt; }
  O[idx] = s / (float)cnt;
}

// ---------------------------------------------------------------------------
extern "C" void kernel_launch(void* const* d_in, const int* in_sizes, int n_in,
                              void* d_out, int out_size, void* d_ws, size_t ws_size,
                              hipStream_t stream)
{
  (void)in_sizes; (void)n_in; (void)out_size; (void)ws_size;
  const float* X    = (const float*)d_in[0];
  const float* W1   = (const float*)d_in[1];
  const float* b1   = (const float*)d_in[2];
  const float* W2   = (const float*)d_in[3];
  const float* b2   = (const float*)d_in[4];
  const float* Wq   = (const float*)d_in[5];
  const float* Wk   = (const float*)d_in[6];
  const float* Wv   = (const float*)d_in[7];
  const float* Wagg = (const float*)d_in[8];
  const float* bagg = (const float*)d_in[9];
  const float* Wdec = (const float*)d_in[10];
  const float* bdec = (const float*)d_in[11];
  float* OUT = (float*)d_out;

  char* w = (char*)d_ws;
  size_t off = 0;
  auto take = [&](size_t bytes) -> void* {
    void* p = (void*)(w + off);
    off += (bytes + 255) & ~(size_t)255;
    return p;
  };

  __bf16* H1    = (__bf16*)take((size_t)Mrows * E1 * 2);  // enc1 output (bf16)
  float*  Xe    = (float*) take((size_t)Mrows * E2 * 4);  // enc2 output
  float*  P     = (float*) take((size_t)NP * E2 * 4);     // per-patch means
  float*  Qb    = (float*) take((size_t)NP * E2 * 4);
  float*  Kb    = (float*) take((size_t)NP * E2 * 4);
  float*  Vb    = (float*) take((size_t)NP * E2 * 4);
  float*  Zb    = (float*) take((size_t)NP * E2 * 4);
  __bf16* W1t   = (__bf16*)take((size_t)Dc * E1 * 2);
  __bf16* W2t   = (__bf16*)take((size_t)E1 * E2 * 2);
  __bf16* Wqt   = (__bf16*)take((size_t)E2 * E2 * 2);
  __bf16* Wkt   = (__bf16*)take((size_t)E2 * E2 * 2);
  __bf16* Wvt   = (__bf16*)take((size_t)E2 * E2 * 2);
  __bf16* Waggt = (__bf16*)take((size_t)E2 * E2 * 2);
  __bf16* Wdect = (__bf16*)take((size_t)E2 * Dc * 2);
  float*  Av    = P;            // P dead after q/k/v GEMMs
  float*  OUTF  = (float*)H1;   // H1 dead after enc2 GEMM (16MB < 33.5MB)

  auto tg = [](int n) { return (n + 255) / 256; };
  cvt_transpose<<<tg(Dc * E1), 256, 0, stream>>>(W1,   W1t,   Dc, E1);
  cvt_transpose<<<tg(E1 * E2), 256, 0, stream>>>(W2,   W2t,   E1, E2);
  cvt_transpose<<<tg(E2 * E2), 256, 0, stream>>>(Wq,   Wqt,   E2, E2);
  cvt_transpose<<<tg(E2 * E2), 256, 0, stream>>>(Wk,   Wkt,   E2, E2);
  cvt_transpose<<<tg(E2 * E2), 256, 0, stream>>>(Wv,   Wvt,   E2, E2);
  cvt_transpose<<<tg(E2 * E2), 256, 0, stream>>>(Wagg, Waggt, E2, E2);
  cvt_transpose<<<tg(E2 * Dc), 256, 0, stream>>>(Wdec, Wdect, E2, Dc);

  auto blocks = [](int mtiles, int n) {        // 16x64 tile per wave, 4/block
    int tiles = mtiles * (n >> 6);
    return (tiles + 3) / 4;
  };

  // enc1: leaky(X @ W1 + b1) -> bf16   [65536 x 512 -> 256]
  gemm_wmma<float, __bf16, 1><<<blocks(Mrows / 16, E1), 128, 0, stream>>>(
      X, W1t, b1, H1, Mrows, E1, Dc);
  // enc2: H1 @ W2 + b2 -> f32          [65536 x 256 -> 64]
  gemm_wmma<__bf16, float, 0><<<blocks(Mrows / 16, E2), 128, 0, stream>>>(
      H1, W2t, b2, Xe, Mrows, E2, E1);

  patch_mean<<<tg(NP * E2), 256, 0, stream>>>(Xe, P);

  const int qkvBlocks = blocks(NP / 16, E2);   // 511 tiles -> 128 blocks
  gemm_wmma<float, float, 0><<<qkvBlocks, 128, 0, stream>>>(P, Wqt, nullptr, Qb, NP, E2, E2);
  gemm_wmma<float, float, 0><<<qkvBlocks, 128, 0, stream>>>(P, Wkt, nullptr, Kb, NP, E2, E2);
  gemm_wmma<float, float, 0><<<qkvBlocks, 128, 0, stream>>>(P, Wvt, nullptr, Vb, NP, E2, E2);

  rownorm<<<tg(NP), 256, 0, stream>>>(Qb, NP);
  rownorm<<<tg(NP), 256, 0, stream>>>(Kb, NP);

  attention_topk<<<NP / 8, 256, 0, stream>>>(Qb, Kb, Vb, Av, NP);

  // agg: leaky(Av @ W_agg + b_agg)     [8176 x 64 -> 64]
  gemm_wmma<float, float, 1><<<qkvBlocks, 128, 0, stream>>>(Av, Waggt, bagg, Zb, NP, E2, E2);

  patch_scatter<<<tg(Mrows * E2), 256, 0, stream>>>(Zb, OUTF);

  // decode: OUTF @ W_dec + b_dec -> d_out   [65536 x 64 -> 512]
  gemm_wmma<float, float, 0><<<blocks(Mrows / 16, Dc), 128, 0, stream>>>(
      OUTF, Wdect, bdec, OUT, Mrows, Dc, E2);
}
